// Attention_83468394430810
// MI455X (gfx1250) — matrix operand order repaired and checked by reading, not compiled, not run on hardware
//
#include <hip/hip_runtime.h>
#include <hip/hip_bf16.h>
#include <math.h>

typedef __attribute__((ext_vector_type(16))) _Float16 v16h;
typedef __attribute__((ext_vector_type(8)))  _Float16 v8h;
typedef __attribute__((ext_vector_type(4)))  _Float16 v4h;
typedef __attribute__((ext_vector_type(8)))  float    v8f;

#define B_    2
#define S_    2048
#define D_    4096
#define NH_   32
#define NL_   8
#define HD_   128
#define QKVN  6144      // (NH + 2*NL) * HD
#define TOK   (B_ * S_) // 4096 tokens

// ---------------------------------------------------------------------------
// Async global -> LDS copy (CDNA5): 16B per lane, tracked by ASYNCcnt.
// VGLOBAL encoding: VDST = per-lane LDS byte offset, VADDR = 64-bit global
// address, saddr = off.  Generic LDS pointers carry the byte offset in the
// low 32 bits (aperture in [63:32]), so truncation yields the DS address.
// ---------------------------------------------------------------------------
static __device__ __forceinline__ void async_copy_b128(void* lds, const void* gptr) {
  uint32_t lds_off = (uint32_t)(uintptr_t)lds;
  uint64_t ga = (uint64_t)(uintptr_t)gptr;
  asm volatile("global_load_async_to_lds_b128 %0, %1, off"
               :: "v"(lds_off), "v"(ga)
               : "memory");
}
static __device__ __forceinline__ void wait_async0() {
  asm volatile("s_wait_asynccnt 0x0" ::: "memory");
}

// ---------------------------------------------------------------------------
// A-fragment loader for V_WMMA_F32_16X16X32_F16.
// Contiguous K order (matches B fragments): lanes 0-15 hold K=[0..15],
// lanes 16-31 hold K=[16..31]. Each lane: two contiguous
// 8-half (16B) chunks at (lh*16) and (lh*16 + 8). p = row base + k0.
// ---------------------------------------------------------------------------
static __device__ __forceinline__ v16h load_a_frag(const _Float16* p, int lh) {
  v8h lo = *(const v8h*)(p + lh * 16);
  v8h hi = *(const v8h*)(p + lh * 16 + 8);
  v16h r;
#pragma unroll
  for (int i = 0; i < 8; ++i) { r[i] = lo[i]; r[i + 8] = hi[i]; }
  return r;
}

// B-fragment (32x16, K x N): lane n holds column n, K=[lh*16 .. lh*16+15],
// contiguous 32B per lane when the matrix is stored [N][K] row-major.

__global__ __launch_bounds__(256) void f32_to_f16_kernel(const float* __restrict__ src,
                                                         _Float16* __restrict__ dst,
                                                         int n) {
  int i = (blockIdx.x * 256 + threadIdx.x) * 4;
  if (i < n) {
    float4 f = *(const float4*)(src + i);
    v4h h = { (_Float16)f.x, (_Float16)f.y, (_Float16)f.z, (_Float16)f.w };
    *(v4h*)(dst + i) = h;
  }
}

// ---------------------------------------------------------------------------
// LDS-staged, double-buffered WMMA GEMM with async global->LDS staging:
//   C[M,N] = A[M,K] (f16 row-major) * W[N,K]^T (f16).
// Block = 256 threads = 8 waves. Block tile 128x128, wave tile 32x64
// (2 M-tiles x 4 N-tiles). K stepped by 32; each step stages 128x32 A and
// 128x32 B tiles into LDS via GLOBAL_LOAD_ASYNC_TO_LDS_B128 (no VGPR round
// trip); the ASYNCcnt wait sits after the 8-WMMA compute block so transfers
// overlap the MMA phase.  LDS rows padded to 40 halves (80B): lane stride 20
// banks -> 16 distinct 4-bank-aligned starts -> conflict-free b128 reads.
// ---------------------------------------------------------------------------
template <bool F32OUT>
__global__ __launch_bounds__(256) void gemm_wmma_kernel(const _Float16* __restrict__ A,
                                                        const _Float16* __restrict__ W,
                                                        void* __restrict__ Cout,
                                                        int M, int N, int K) {
  constexpr int LSTR = 40;  // padded LDS row stride in halves
  __shared__ _Float16 As[2][128 * LSTR];
  __shared__ _Float16 Bs[2][128 * LSTR];

  const int tid  = threadIdx.x;
  const int wave = tid >> 5;
  const int lane = tid & 31;
  const int lr = lane & 15;   // row (A) / col (W) within a fragment
  const int lh = lane >> 4;   // K-half selector
  const int wm = wave >> 1;   // 0..3
  const int wn = wave & 1;    // 0..1
  const int blockM = blockIdx.y * 128;
  const int blockN = blockIdx.x * 128;

  // Stage-loader assignment: thread t handles row (t>>1), 16-half piece (t&1)
  const int ldRow = tid >> 1;
  const int ldOff = (tid & 1) * 16;
  const _Float16* Ag = A + (size_t)(blockM + ldRow) * K + ldOff;
  const _Float16* Bg = W + (size_t)(blockN + ldRow) * K + ldOff;
  const int lds_w = ldRow * LSTR + ldOff;

  // prologue: stage k-step 0 into buffer 0 (async, 16B x4 per thread)
  async_copy_b128(&As[0][lds_w],     Ag);
  async_copy_b128(&As[0][lds_w + 8], Ag + 8);
  async_copy_b128(&Bs[0][lds_w],     Bg);
  async_copy_b128(&Bs[0][lds_w + 8], Bg + 8);
  wait_async0();
  __syncthreads();

  v8f acc[2][4];
#pragma unroll
  for (int i = 0; i < 2; ++i)
#pragma unroll
    for (int j = 0; j < 4; ++j) acc[i][j] = (v8f)0.0f;

  const int nk = K >> 5;
  for (int ks = 0; ks < nk; ++ks) {
    const int buf = ks & 1;
    const bool more = (ks + 1) < nk;
    if (more) {                          // issue next tile's async copies early
      const _Float16* ap = Ag + (ks + 1) * 32;
      const _Float16* bp = Bg + (ks + 1) * 32;
      const int nb = buf ^ 1;
      async_copy_b128(&As[nb][lds_w],     ap);
      async_copy_b128(&As[nb][lds_w + 8], ap + 8);
      async_copy_b128(&Bs[nb][lds_w],     bp);
      async_copy_b128(&Bs[nb][lds_w + 8], bp + 8);
    }

    v16h a[2], b[4];
#pragma unroll
    for (int i = 0; i < 2; ++i)
      a[i] = load_a_frag(&As[buf][(wm * 32 + i * 16 + lr) * LSTR], lh);
#pragma unroll
    for (int j = 0; j < 4; ++j)
      b[j] = *(const v16h*)(&Bs[buf][(wn * 64 + j * 16 + lr) * LSTR + lh * 16]);
#pragma unroll
    for (int i = 0; i < 2; ++i)
#pragma unroll
      for (int j = 0; j < 4; ++j)
        acc[i][j] = __builtin_amdgcn_wmma_f32_16x16x32_f16(
            false, a[i], false, b[j], (short)0, acc[i][j], false, false);

    if (more) wait_async0();             // next buffer landed during the MMAs
    __syncthreads();
  }

  // C/D layout: VGPR r -> lanes 0-15: (M=r, N=lane); lanes 16-31: (M=8+r, N=lane-16)
  const int rbase = lh * 8;
  const int tileM = blockM + wm * 32;
  const int tileN = blockN + wn * 64;
#pragma unroll
  for (int i = 0; i < 2; ++i)
#pragma unroll
    for (int j = 0; j < 4; ++j)
#pragma unroll
      for (int r = 0; r < 8; ++r) {
        size_t row = (size_t)(tileM + i * 16 + rbase + r);
        size_t col = (size_t)(tileN + j * 16 + lr);
        if (F32OUT)
          ((float*)Cout)[row * N + col] = acc[i][j][r];
        else
          ((_Float16*)Cout)[row * N + col] = (_Float16)acc[i][j][r];
      }
}

// ---------------------------------------------------------------------------
// RoPE + repack: qkv[t, 6144] -> Qh[B,NH,S,HD], Kh[B,NL,S,HD], Vh[B,NL,S,HD]
// One thread per channel-pair (3072 pairs per token).
// ---------------------------------------------------------------------------
__global__ __launch_bounds__(256) void rope_pack_kernel(const _Float16* __restrict__ qkv,
                                                        const float* __restrict__ freqs,
                                                        _Float16* __restrict__ Qh,
                                                        _Float16* __restrict__ Kh,
                                                        _Float16* __restrict__ Vh) {
  int idx = blockIdx.x * 256 + threadIdx.x;      // [0, B*S*3072)
  int c = idx % 3072;
  int t = idx / 3072;
  int s = t & (S_ - 1);
  int b = t >> 11;
  const _Float16* row = qkv + (size_t)t * QKVN;
  if (c < 2048) {                                // Q
    int h = c >> 6, dp = c & 63;
    float x0 = (float)row[h * HD_ + 2 * dp];
    float x1 = (float)row[h * HD_ + 2 * dp + 1];
    float cs = freqs[s * HD_ + dp * 2];
    float sn = freqs[s * HD_ + dp * 2 + 1];
    size_t o = (((size_t)b * NH_ + h) * S_ + s) * HD_ + 2 * dp;
    Qh[o]     = (_Float16)(x0 * cs - x1 * sn);
    Qh[o + 1] = (_Float16)(x1 * cs + x0 * sn);
  } else if (c < 2560) {                         // K
    int l = (c - 2048) >> 6, dp = (c - 2048) & 63;
    float x0 = (float)row[D_ + l * HD_ + 2 * dp];
    float x1 = (float)row[D_ + l * HD_ + 2 * dp + 1];
    float cs = freqs[s * HD_ + dp * 2];
    float sn = freqs[s * HD_ + dp * 2 + 1];
    size_t o = (((size_t)b * NL_ + l) * S_ + s) * HD_ + 2 * dp;
    Kh[o]     = (_Float16)(x0 * cs - x1 * sn);
    Kh[o + 1] = (_Float16)(x1 * cs + x0 * sn);
  } else {                                       // V (no rope)
    int l = (c - 2560) >> 6, dp = (c - 2560) & 63;
    size_t o = (((size_t)b * NL_ + l) * S_ + s) * HD_ + 2 * dp;
    Vh[o]     = row[D_ + 1024 + l * HD_ + 2 * dp];
    Vh[o + 1] = row[D_ + 1024 + l * HD_ + 2 * dp + 1];
  }
}

// ---------------------------------------------------------------------------
// Causal flash attention (online softmax), GQA with group = 4.
// Grid: (S/64, NH, B).  Block: 128 threads = 4 waves; wave w owns query rows
// [qblk*64 + w*16, +16).  K tile (64x128) staged via async LDS copies; V tile
// transposed through VGPRs; P goes through per-wave LDS for C->A layout.
// Output Yh is [B, S, NH, HD] f16 (== [token, 4096] for the final GEMM).
// ---------------------------------------------------------------------------
__global__ __launch_bounds__(128) void flash_attn_kernel(const _Float16* __restrict__ Qh,
                                                         const _Float16* __restrict__ Kh,
                                                         const _Float16* __restrict__ Vh,
                                                         _Float16* __restrict__ Yh) {
  __shared__ _Float16 Ks[64 * HD_];       // [key][d]          16 KB
  __shared__ _Float16 Vt[HD_ * 64];       // [d][key]          16 KB
  __shared__ _Float16 Ps[4 * 16 * 64];    // per-wave P tile     8 KB

  const int tid  = threadIdx.x;
  const int wave = tid >> 5;
  const int lane = tid & 31;
  const int lr = lane & 15;
  const int lh = lane >> 4;

  const int qblk = blockIdx.x;            // 64-row query block
  const int h    = blockIdx.y;
  const int b    = blockIdx.z;
  const int hkv  = h >> 2;                // GQA: NH/NL = 4

  const _Float16* Qg = Qh + (((size_t)b * NH_ + h) * S_) * HD_;
  const _Float16* Kg = Kh + (((size_t)b * NL_ + hkv) * S_) * HD_;
  const _Float16* Vg = Vh + (((size_t)b * NL_ + hkv) * S_) * HD_;

  const int q0 = qblk * 64 + wave * 16;   // first query row of this wave
  const int qrow_base = q0 + lh * 8;      // M rows this lane sees (+r)

  // Q fragments: 16 rows x 128 cols = 4 A-frags (K steps of 32)
  v16h aq[4];
#pragma unroll
  for (int kk = 0; kk < 4; ++kk)
    aq[kk] = load_a_frag(Qg + (size_t)(q0 + lr) * HD_ + kk * 32, lh);

  v8f o[8];
#pragma unroll
  for (int nt = 0; nt < 8; ++nt) o[nt] = (v8f)0.0f;
  float mrow[8], lrow[8];
#pragma unroll
  for (int r = 0; r < 8; ++r) { mrow[r] = -INFINITY; lrow[r] = 0.0f; }

  const float scale = 0.08838834764831845f;  // 1/sqrt(128)
  _Float16* pw = Ps + wave * 16 * 64;

  for (int kb = 0; kb <= qblk; ++kb) {
    // ---- LDS staging: K async (straight copy), V via regs (transposed) ----
#pragma unroll
    for (int i = 0; i < 8; ++i) {
      int c = i * 128 + tid;              // 16B chunk id, 1024 total
      async_copy_b128(&((v8h*)Ks)[c], Kg + (size_t)(kb * 64) * HD_ + c * 8);
      int krow = c >> 4, col8 = (c & 15) * 8;
      v8h vv = *(const v8h*)(Vg + (size_t)(kb * 64 + krow) * HD_ + col8);
#pragma unroll
      for (int e = 0; e < 8; ++e) Vt[(col8 + e) * 64 + krow] = vv[e];
    }
    wait_async0();
    __syncthreads();

    // ---- S = Q K^T : 16x64 per wave ----
    v8f sa[4];
#pragma unroll
    for (int nt = 0; nt < 4; ++nt) sa[nt] = (v8f)0.0f;
#pragma unroll
    for (int kk = 0; kk < 4; ++kk) {
#pragma unroll
      for (int nt = 0; nt < 4; ++nt) {
        v16h bk = *(const v16h*)(Ks + (nt * 16 + lr) * HD_ + kk * 32 + lh * 16);
        sa[nt] = __builtin_amdgcn_wmma_f32_16x16x32_f16(
            false, aq[kk], false, bk, (short)0, sa[nt], false, false);
      }
    }

    // ---- scale + causal mask ----
#pragma unroll
    for (int nt = 0; nt < 4; ++nt)
#pragma unroll
      for (int r = 0; r < 8; ++r) {
        float v = sa[nt][r] * scale;
        int qi = qrow_base + r;
        int ki = kb * 64 + nt * 16 + lr;
        sa[nt][r] = (ki <= qi) ? v : -3.0e38f;
      }

    // ---- online softmax (row stats across 16 lanes sharing an M row) ----
    float alpha[8];
#pragma unroll
    for (int r = 0; r < 8; ++r) {
      float mx = fmaxf(fmaxf(sa[0][r], sa[1][r]), fmaxf(sa[2][r], sa[3][r]));
#pragma unroll
      for (int off = 8; off >= 1; off >>= 1) mx = fmaxf(mx, __shfl_xor(mx, off, 16));
      float mnew = fmaxf(mrow[r], mx);
      alpha[r] = (mrow[r] == -INFINITY) ? 0.0f : __expf(mrow[r] - mnew);
      mrow[r] = mnew;
      float rs = 0.0f;
#pragma unroll
      for (int nt = 0; nt < 4; ++nt) {
        float p = __expf(sa[nt][r] - mnew);
        sa[nt][r] = p;
        rs += p;
      }
#pragma unroll
      for (int off = 8; off >= 1; off >>= 1) rs += __shfl_xor(rs, off, 16);
      lrow[r] = lrow[r] * alpha[r] + rs;
    }
#pragma unroll
    for (int nt = 0; nt < 8; ++nt)
#pragma unroll
      for (int r = 0; r < 8; ++r) o[nt][r] *= alpha[r];

    // ---- P: C-layout regs -> LDS -> A-layout frags (same-wave DS, in order) ----
#pragma unroll
    for (int nt = 0; nt < 4; ++nt)
#pragma unroll
      for (int r = 0; r < 8; ++r)
        pw[(lh * 8 + r) * 64 + nt * 16 + lr] = (_Float16)sa[nt][r];

    // ---- O += P V : 16x128, K = 64 in 2 steps ----
#pragma unroll
    for (int kk = 0; kk < 2; ++kk) {
      v16h ap = load_a_frag(pw + lr * 64 + kk * 32, lh);
#pragma unroll
      for (int nt = 0; nt < 8; ++nt) {
        v16h bv = *(const v16h*)(Vt + (nt * 16 + lr) * 64 + kk * 32 + lh * 16);
        o[nt] = __builtin_amdgcn_wmma_f32_16x16x32_f16(
            false, ap, false, bv, (short)0, o[nt], false, false);
      }
    }
    __syncthreads();
  }

  // ---- normalize and write Yh[b, s, h, d] ----
#pragma unroll
  for (int r = 0; r < 8; ++r) {
    float inv = 1.0f / lrow[r];
    int s = qrow_base + r;
#pragma unroll
    for (int nt = 0; nt < 8; ++nt) {
      size_t oidx = (((size_t)b * S_ + s) * NH_ + h) * HD_ + nt * 16 + lr;
      Yh[oidx] = (_Float16)(o[nt][r] * inv);
    }
  }
}

// ---------------------------------------------------------------------------
// Host launcher
// ---------------------------------------------------------------------------
extern "C" void kernel_launch(void* const* d_in, const int* in_sizes, int n_in,
                              void* d_out, int out_size, void* d_ws, size_t ws_size,
                              hipStream_t stream) {
  const float* x     = (const float*)d_in[0];   // [B,S,D]
  const float* freqs = (const float*)d_in[1];   // [S,HD/2,2]
  // d_in[2] = causal mask (bool) -- applied analytically, not read
  const float* wqkv  = (const float*)d_in[3];   // [6144,4096]
  const float* wo    = (const float*)d_in[4];   // [4096,4096]

  char* ws = (char*)d_ws;
  size_t off = 0;
  _Float16* x_h   = (_Float16*)(ws + off); off += (size_t)TOK * D_ * 2;       // 32 MB
  _Float16* w1_h  = (_Float16*)(ws + off); off += (size_t)QKVN * D_ * 2;      // 48 MB
  _Float16* w2_h  = (_Float16*)(ws + off); off += (size_t)D_ * D_ * 2;        // 32 MB
  _Float16* qkv_h = (_Float16*)(ws + off); off += (size_t)TOK * QKVN * 2;     // 48 MB
  _Float16* Qh    = (_Float16*)(ws + off); off += (size_t)B_ * NH_ * S_ * HD_ * 2;
  _Float16* Kh    = (_Float16*)(ws + off); off += (size_t)B_ * NL_ * S_ * HD_ * 2;
  _Float16* Vh    = (_Float16*)(ws + off); off += (size_t)B_ * NL_ * S_ * HD_ * 2;
  _Float16* Yh    = (_Float16*)(ws + off); off += (size_t)TOK * D_ * 2;

  // 1) downconvert inputs to f16 (4 elems/thread, float4 in / v4h out)
  {
    int n;
    n = TOK * D_;
    f32_to_f16_kernel<<<(n / 4 + 255) / 256, 256, 0, stream>>>(x, x_h, n);
    n = QKVN * D_;
    f32_to_f16_kernel<<<(n / 4 + 255) / 256, 256, 0, stream>>>(wqkv, w1_h, n);
    n = D_ * D_;
    f32_to_f16_kernel<<<(n / 4 + 255) / 256, 256, 0, stream>>>(wo, w2_h, n);
  }

  // 2) qkv = x @ wqkv^T   (M=4096, N=6144, K=4096)
  gemm_wmma_kernel<false><<<dim3(QKVN / 128, TOK / 128), 256, 0, stream>>>(
      x_h, w1_h, (void*)qkv_h, TOK, QKVN, D_);

  // 3) RoPE + repack
  {
    int n = TOK * 3072;
    rope_pack_kernel<<<n / 256, 256, 0, stream>>>(qkv_h, freqs, Qh, Kh, Vh);
  }

  // 4) causal flash attention
  flash_attn_kernel<<<dim3(S_ / 64, NH_, B_), 128, 0, stream>>>(Qh, Kh, Vh, Yh);

  // 5) out = y_attn @ wo^T  (M=4096, N=4096, K=4096), f32 output
  gemm_wmma_kernel<true><<<dim3(D_ / 128, TOK / 128), 256, 0, stream>>>(
      Yh, w2_h, d_out, TOK, D_, D_);

  (void)in_sizes; (void)n_in; (void)out_size; (void)ws_size;
}
